// GraphAttentionLayer_52544629899820
// MI455X (gfx1250) — matrix operand order repaired and checked by reading, not compiled, not run on hardware
//
#include <hip/hip_runtime.h>

typedef __attribute__((ext_vector_type(16))) __bf16 v16bf;
typedef __attribute__((ext_vector_type(8)))  __bf16 v8bf;
typedef __attribute__((ext_vector_type(8)))  float  v8f;

#define WMMA_BF16(a, b, c) \
  __builtin_amdgcn_wmma_f32_16x16x32_bf16(false, (a), false, (b), (short)0, (c), false, false)

__device__ __forceinline__ v8f zerov8() {
  v8f r;
#pragma unroll
  for (int i = 0; i < 8; ++i) r[i] = 0.f;
  return r;
}

// A-matrix fragment (16x32 bf16): lane half hf reads k = [hf*8, hf*8+7] and [16+hf*8, 16+hf*8+7]
__device__ __forceinline__ v16bf frag_a(const __bf16* p) {
  v8bf lo = *(const v8bf*)(p);
  v8bf hi = *(const v8bf*)(p + 16);
  v16bf r;
#pragma unroll
  for (int t = 0; t < 8; ++t) { r[t] = lo[t]; r[t + 8] = hi[t]; }
  return r;
}

// B-matrix fragment (32x16 bf16): lane half hf reads k = [hf*8, hf*8+7] and [16+hf*8, 16+hf*8+7]
__device__ __forceinline__ v16bf frag_b(const __bf16* p) {
  v8bf lo = *(const v8bf*)(p);
  v8bf hi = *(const v8bf*)(p + 16);
  v16bf r;
#pragma unroll
  for (int t = 0; t < 8; ++t) { r[t] = lo[t]; r[t + 8] = hi[t]; }
  return r;
}

// Async 32B global->LDS copy (two b128 ops; offset applies to BOTH addresses).
__device__ __forceinline__ void async_copy32(__bf16* lds_dst, const __bf16* gsrc) {
  unsigned ldsa = (unsigned)(size_t)lds_dst;
  unsigned long long ga = (unsigned long long)(size_t)gsrc;
  asm volatile("global_load_async_to_lds_b128 %0, %1, off"
               :: "v"(ldsa), "v"(ga) : "memory");
  asm volatile("global_load_async_to_lds_b128 %0, %1, off offset:16"
               :: "v"(ldsa), "v"(ga) : "memory");
}

// ---------------------------------------------------------------------------
// Kernel 1: QKV projection GEMM.  C[8192x512] = x[8192x512] @ W[512x512] + b
// blockIdx.z selects Q/K/V.  Output bf16 in (B,H,N,DH) layout.
// block = 256 threads (8 waves), tile 128(M) x 64(N), K-step 32.
// ---------------------------------------------------------------------------
__global__ __launch_bounds__(256) void qkv_gemm_kernel(
    const float* __restrict__ x,
    const float* __restrict__ Wq, const float* __restrict__ bq,
    const float* __restrict__ Wk, const float* __restrict__ bk,
    const float* __restrict__ Wv, const float* __restrict__ bv,
    __bf16* __restrict__ qb, __bf16* __restrict__ kb, __bf16* __restrict__ vb) {
  constexpr int SA = 40;  // lA row stride (bf16), 80B: 16B-aligned, conflict-free
  constexpr int SB = 40;
  __shared__ __attribute__((aligned(16))) __bf16 lA[128 * SA];  // [m][k]
  __shared__ __attribute__((aligned(16))) __bf16 lB[64 * SB];   // [c][k] (W transposed)

  const int tid = threadIdx.x;
  const int w = tid >> 5;
  const int lane = tid & 31;
  const int ln = lane & 15, hf = lane >> 4;
  const int m0 = blockIdx.y * 128;
  const int c0 = blockIdx.x * 64;
  const int z = blockIdx.z;

  const float* W    = (z == 0) ? Wq : (z == 1) ? Wk : Wv;
  const float* bias = (z == 0) ? bq : (z == 1) ? bk : bv;
  __bf16* out       = (z == 0) ? qb : (z == 1) ? kb : vb;

  const int wm = w >> 1, wn = w & 1;  // 4x2 wave grid, 32x32 per wave

  v8f acc[2][2];
#pragma unroll
  for (int i = 0; i < 2; ++i)
#pragma unroll
    for (int j = 0; j < 2; ++j) acc[i][j] = zerov8();

  const int ar = tid >> 1, akh = (tid & 1) * 16;  // A staging: row, k-half
  const int bkr = tid >> 3, bcq = (tid & 7) * 8;  // B staging: k, c-quad

  for (int kk = 0; kk < 512; kk += 32) {
    __syncthreads();
    {
      const float* src = x + (m0 + ar) * 512 + kk + akh;
      __bf16* dst = &lA[ar * SA + akh];
#pragma unroll
      for (int j = 0; j < 16; ++j) dst[j] = (__bf16)src[j];
      const float* wsrc = W + (kk + bkr) * 512 + c0 + bcq;
#pragma unroll
      for (int j = 0; j < 8; ++j) lB[(bcq + j) * SB + bkr] = (__bf16)wsrc[j];
    }
    __syncthreads();

    v16bf a0 = frag_a(&lA[(wm * 32 + 0  + ln) * SA + hf * 8]);
    v16bf a1 = frag_a(&lA[(wm * 32 + 16 + ln) * SA + hf * 8]);
    v16bf b0 = frag_b(&lB[(wn * 32 + 0  + ln) * SB + hf * 8]);
    v16bf b1 = frag_b(&lB[(wn * 32 + 16 + ln) * SB + hf * 8]);
    acc[0][0] = WMMA_BF16(a0, b0, acc[0][0]);
    acc[0][1] = WMMA_BF16(a0, b1, acc[0][1]);
    acc[1][0] = WMMA_BF16(a1, b0, acc[1][0]);
    acc[1][1] = WMMA_BF16(a1, b1, acc[1][1]);
  }

#pragma unroll
  for (int mi = 0; mi < 2; ++mi)
#pragma unroll
    for (int ni = 0; ni < 2; ++ni) {
      const int col = c0 + wn * 32 + ni * 16 + ln;
      const float bc = bias[col];
      const int hh = col >> 6, dh = col & 63;
#pragma unroll
      for (int i = 0; i < 8; ++i) {
        const int m = m0 + wm * 32 + mi * 16 + i + 8 * hf;
        const int bb = m >> 10, nn = m & 1023;
        out[(((bb << 3) + hh) * 1024 + nn) * 64 + dh] = (__bf16)(acc[mi][ni][i] + bc);
      }
    }
}

// ---------------------------------------------------------------------------
// Kernel 2: flash attention per (b,h).  Each wave owns 16 query rows
// (block handles 128); loops over 16 key blocks of 64.  Online softmax.
// K tile staged with async global->LDS DMA; V tile transposed through VGPRs.
// ---------------------------------------------------------------------------
__global__ __launch_bounds__(256) void attn_kernel(
    const __bf16* __restrict__ qb, const __bf16* __restrict__ kb,
    const __bf16* __restrict__ vb, const float* __restrict__ pos_bias,
    __bf16* __restrict__ attn_out) {
  constexpr int ST = 72;  // 144B row stride: 16B-aligned, conflict-free
  __shared__ __attribute__((aligned(16))) __bf16 lK[64 * ST];      // [j][d]
  __shared__ __attribute__((aligned(16))) __bf16 lV[64 * ST];      // [d][j] (transposed)
  __shared__ __attribute__((aligned(16))) __bf16 lP[8 * 16 * ST];  // per-wave P tiles

  const int tid = threadIdx.x;
  const int w = tid >> 5;
  const int lane = tid & 31;
  const int ln = lane & 15, hf = lane >> 4;
  const int bh = blockIdx.y;  // b*8 + h
  const int b = bh >> 3, h = bh & 7;
  const int q0 = blockIdx.x * 128 + w * 16;

  // Q fragments, pre-scaled by 1/sqrt(64) = 0.125 (exact in bf16)
  const __bf16* qrow = qb + ((size_t)bh * 1024 + q0 + ln) * 64;
  v16bf aq[2];
#pragma unroll
  for (int c = 0; c < 2; ++c) {
    aq[c] = frag_a(qrow + c * 32 + hf * 8);
#pragma unroll
    for (int t = 0; t < 16; ++t) aq[c][t] = (__bf16)((float)aq[c][t] * 0.125f);
  }

  v8f o[4];
#pragma unroll
  for (int dt = 0; dt < 4; ++dt) o[dt] = zerov8();
  float mrow[8], lrow[8];
#pragma unroll
  for (int i = 0; i < 8; ++i) { mrow[i] = -3.0e38f; lrow[i] = 0.f; }

  const int sj = tid >> 2;        // staging: key row 0..63
  const int sd = (tid & 3) * 16;  // staging: d quarter
  const __bf16* kbase = kb + (size_t)bh * 1024 * 64;
  const __bf16* vbase = vb + (size_t)bh * 1024 * 64;
  __bf16* pw = &lP[w * 16 * ST];

  for (int j0 = 0; j0 < 1024; j0 += 64) {
    __syncthreads();
    {
      // K tile: raw bf16 copy -> async DMA to LDS (no VGPR round-trip)
      async_copy32(&lK[sj * ST + sd], kbase + (j0 + sj) * 64 + sd);
      // V tile: transpose through VGPRs (scatter b16 stores)
      const __bf16* vsrc = vbase + (j0 + sj) * 64 + sd;
#pragma unroll
      for (int j = 0; j < 16; ++j) lV[(sd + j) * ST + sj] = vsrc[j];
      asm volatile("s_wait_asynccnt 0x0" ::: "memory");
    }
    __syncthreads();

    // S(16x64) = Qs @ K^T
    v8f s[4];
#pragma unroll
    for (int jt = 0; jt < 4; ++jt) s[jt] = zerov8();
#pragma unroll
    for (int c = 0; c < 2; ++c)
#pragma unroll
      for (int jt = 0; jt < 4; ++jt) {
        v16bf bkf = frag_b(&lK[(jt * 16 + ln) * ST + c * 32 + hf * 8]);
        s[jt] = WMMA_BF16(aq[c], bkf, s[jt]);
      }

    // + positional bias (prefetch next block's slice while we compute)
    const float* pb = pos_bias + ((size_t)h * 1024 + q0 + 8 * hf) * 1024 + j0 + ln;
    if (j0 + 64 < 1024) __builtin_prefetch(pb + 64, 0, 1);
    float sv[4][8];
#pragma unroll
    for (int jt = 0; jt < 4; ++jt)
#pragma unroll
      for (int i = 0; i < 8; ++i) sv[jt][i] = s[jt][i] + pb[i * 1024 + jt * 16];

    // online softmax (row reductions across 16-lane column groups)
#pragma unroll
    for (int i = 0; i < 8; ++i) {
      float mb = fmaxf(fmaxf(sv[0][i], sv[1][i]), fmaxf(sv[2][i], sv[3][i]));
      mb = fmaxf(mb, __shfl_xor(mb, 1));
      mb = fmaxf(mb, __shfl_xor(mb, 2));
      mb = fmaxf(mb, __shfl_xor(mb, 4));
      mb = fmaxf(mb, __shfl_xor(mb, 8));
      const float mn = fmaxf(mrow[i], mb);
      const float alpha = __expf(mrow[i] - mn);
      mrow[i] = mn;
      float rs = 0.f;
#pragma unroll
      for (int jt = 0; jt < 4; ++jt) {
        const float pv = __expf(sv[jt][i] - mn);
        sv[jt][i] = pv;
        rs += pv;
      }
      rs += __shfl_xor(rs, 1);
      rs += __shfl_xor(rs, 2);
      rs += __shfl_xor(rs, 4);
      rs += __shfl_xor(rs, 8);
      lrow[i] = lrow[i] * alpha + rs;
#pragma unroll
      for (int dt = 0; dt < 4; ++dt) o[dt][i] *= alpha;
    }

    // transpose P through per-wave LDS tile (cross-lane, DS pipe is in-order)
#pragma unroll
    for (int jt = 0; jt < 4; ++jt)
#pragma unroll
      for (int i = 0; i < 8; ++i)
        pw[(i + 8 * hf) * ST + jt * 16 + ln] = (__bf16)sv[jt][i];
    asm volatile("s_wait_dscnt 0x0" ::: "memory");

    // O(16x64) += P @ V
#pragma unroll
    for (int c = 0; c < 2; ++c) {
      v16bf ap = frag_a(&pw[ln * ST + c * 32 + hf * 8]);
#pragma unroll
      for (int dt = 0; dt < 4; ++dt) {
        v16bf bvf = frag_b(&lV[(dt * 16 + ln) * ST + c * 32 + hf * 8]);
        o[dt] = WMMA_BF16(ap, bvf, o[dt]);
      }
    }
  }

  // normalize and store concat-head layout (B*N, 512) bf16
#pragma unroll
  for (int i = 0; i < 8; ++i) {
    const float inv = 1.f / lrow[i];
    const int row = q0 + i + 8 * hf;
    __bf16* dst = attn_out + ((size_t)b * 1024 + row) * 512 + h * 64;
#pragma unroll
    for (int dt = 0; dt < 4; ++dt) dst[dt * 16 + ln] = (__bf16)(o[dt][i] * inv);
  }
}

// ---------------------------------------------------------------------------
// Kernel 3: output projection.  out[8192x512] = attn(bf16) @ Wo + bo  (f32)
// A tile staged with async global->LDS DMA (raw bf16 copy).
// ---------------------------------------------------------------------------
__global__ __launch_bounds__(256) void out_gemm_kernel(
    const __bf16* __restrict__ attn, const float* __restrict__ Wo,
    const float* __restrict__ bo, float* __restrict__ out) {
  constexpr int SA = 40;
  constexpr int SB = 40;
  __shared__ __attribute__((aligned(16))) __bf16 lA[128 * SA];
  __shared__ __attribute__((aligned(16))) __bf16 lB[64 * SB];

  const int tid = threadIdx.x;
  const int w = tid >> 5;
  const int lane = tid & 31;
  const int ln = lane & 15, hf = lane >> 4;
  const int m0 = blockIdx.y * 128;
  const int c0 = blockIdx.x * 64;
  const int wm = w >> 1, wn = w & 1;

  v8f acc[2][2];
#pragma unroll
  for (int i = 0; i < 2; ++i)
#pragma unroll
    for (int j = 0; j < 2; ++j) acc[i][j] = zerov8();

  const int ar = tid >> 1, akh = (tid & 1) * 16;
  const int bkr = tid >> 3, bcq = (tid & 7) * 8;

  for (int kk = 0; kk < 512; kk += 32) {
    __syncthreads();
    {
      async_copy32(&lA[ar * SA + akh], attn + (m0 + ar) * 512 + kk + akh);
      const float* wsrc = Wo + (kk + bkr) * 512 + c0 + bcq;
#pragma unroll
      for (int j = 0; j < 8; ++j) lB[(bcq + j) * SB + bkr] = (__bf16)wsrc[j];
      asm volatile("s_wait_asynccnt 0x0" ::: "memory");
    }
    __syncthreads();

    v16bf a0 = frag_a(&lA[(wm * 32 + 0  + ln) * SA + hf * 8]);
    v16bf a1 = frag_a(&lA[(wm * 32 + 16 + ln) * SA + hf * 8]);
    v16bf b0 = frag_b(&lB[(wn * 32 + 0  + ln) * SB + hf * 8]);
    v16bf b1 = frag_b(&lB[(wn * 32 + 16 + ln) * SB + hf * 8]);
    acc[0][0] = WMMA_BF16(a0, b0, acc[0][0]);
    acc[0][1] = WMMA_BF16(a0, b1, acc[0][1]);
    acc[1][0] = WMMA_BF16(a1, b0, acc[1][0]);
    acc[1][1] = WMMA_BF16(a1, b1, acc[1][1]);
  }

#pragma unroll
  for (int mi = 0; mi < 2; ++mi)
#pragma unroll
    for (int ni = 0; ni < 2; ++ni) {
      const int col = c0 + wn * 32 + ni * 16 + ln;
      const float bc = bo[col];
#pragma unroll
      for (int i = 0; i < 8; ++i) {
        const int m = m0 + wm * 32 + mi * 16 + i + 8 * hf;
        out[(size_t)m * 512 + col] = acc[mi][ni][i] + bc;
      }
    }
}

// ---------------------------------------------------------------------------
extern "C" void kernel_launch(void* const* d_in, const int* in_sizes, int n_in,
                              void* d_out, int out_size, void* d_ws, size_t ws_size,
                              hipStream_t stream) {
  (void)in_sizes; (void)n_in; (void)out_size; (void)ws_size;
  const float* x        = (const float*)d_in[0];
  const float* Wq       = (const float*)d_in[1];
  const float* bq       = (const float*)d_in[2];
  const float* Wk       = (const float*)d_in[3];
  const float* bk       = (const float*)d_in[4];
  const float* Wv       = (const float*)d_in[5];
  const float* bv       = (const float*)d_in[6];
  const float* Wo       = (const float*)d_in[7];
  const float* bo       = (const float*)d_in[8];
  const float* pos_bias = (const float*)d_in[9];
  float* out = (float*)d_out;

  const size_t QKV = (size_t)8 * 8 * 1024 * 64;  // 4M bf16 elements each
  __bf16* qb   = (__bf16*)d_ws;
  __bf16* kb   = qb + QKV;
  __bf16* vb   = kb + QKV;
  __bf16* attn = vb + QKV;  // (B*N, 512) bf16

  qkv_gemm_kernel<<<dim3(8, 64, 3), 256, 0, stream>>>(x, Wq, bq, Wk, bk, Wv, bv, qb, kb, vb);
  attn_kernel<<<dim3(8, 64), 256, 0, stream>>>(qb, kb, vb, pos_bias, attn);
  out_gemm_kernel<<<dim3(8, 64), 256, 0, stream>>>(attn, Wo, bo, out);
}
